// WitnessEncoder_16518444220438
// MI455X (gfx1250) — matrix operand order repaired and checked by reading, not compiled, not run on hardware
//
#include <hip/hip_runtime.h>
#include <hip/hip_bf16.h>

#define BATCH  8
#define NSEQ   2048
#define EMB    256
#define HID    512
#define BN     (BATCH * NSEQ)

typedef __bf16 bf16_t;
typedef __attribute__((ext_vector_type(16))) __bf16 v16bf;
typedef __attribute__((ext_vector_type(2)))  __bf16 v2bf;
typedef __attribute__((ext_vector_type(8)))  float  v8f;
typedef __attribute__((ext_vector_type(4)))  unsigned int v4u;
typedef __attribute__((ext_vector_type(8)))  int v8i;
typedef __attribute__((ext_vector_type(4)))  int v4i;

__device__ __forceinline__ bf16_t to_bf16(float f) { return (bf16_t)f; }

// ---------------------------------------------------------------------------
// Fragment loaders per CDNA5 ISA layouts (wave32, bf16 16x16x32).
// A (16x32): lane L -> M=L&15 ; VGPR v -> K = (v>>2)*16 + (L>>4)*8 + (v&3)*2 {+0,+1}
//            (v=0..3 and v=4..7 are each 16 contiguous bytes -> b128 loads)
// B (32x16): lane L -> N=L&15 ; elements 0..7 -> K = (L>>4)*8 + 0..7, 8..15 -> K = 16 + (L>>4)*8 + 0..7
// C/D:       lane L -> N=L&15 ; VGPR i -> M = i + 8*(L>>4)
// ---------------------------------------------------------------------------

__device__ __forceinline__ v16bf load_a_global(const bf16_t* __restrict__ src,
                                               int ld, int row0, int k0, int lane) {
  const int m = lane & 15, hi = lane >> 4;
  const bf16_t* rp = src + (size_t)(row0 + m) * ld;
  v16bf a;
#pragma unroll
  for (int v = 0; v < 8; ++v) {
    const int k = k0 + (v >> 2) * 16 + hi * 8 + (v & 3) * 2;
    v2bf p = *(const v2bf*)(rp + k);
    a[2 * v]     = p[0];
    a[2 * v + 1] = p[1];
  }
  return a;
}

template <int LD>
__device__ __forceinline__ v16bf load_a_lds(const bf16_t (*ldsA)[LD], int k0, int lane) {
  const int m = lane & 15, hi = lane >> 4;
  v16bf a;
#pragma unroll
  for (int v = 0; v < 8; ++v) {
    const int k = k0 + (v >> 2) * 16 + hi * 8 + (v & 3) * 2;
    v2bf p = *(const v2bf*)(&ldsA[m][k]);
    a[2 * v]     = p[0];
    a[2 * v + 1] = p[1];
  }
  return a;
}

// B fragment: element (k, n) must live at src[n*ld + k] (K contiguous per column).
__device__ __forceinline__ v16bf load_b_global(const bf16_t* __restrict__ src,
                                               int ld, int n0, int k0, int lane) {
  const int n = lane & 15, hi = lane >> 4;
  typedef __attribute__((ext_vector_type(8))) __bf16 v8bf;
  const bf16_t* p = src + (size_t)(n0 + n) * ld + k0 + hi * 8;
  v8bf lo8 = *(const v8bf*)p, hi8 = *(const v8bf*)(p + 16);
  return __builtin_shufflevector(lo8, hi8, 0,1,2,3,4,5,6,7,8,9,10,11,12,13,14,15);
}

__device__ __forceinline__ v16bf load_b_lds(const bf16_t (*ldsB)[HID], int k0, int lane) {
  const int n = lane & 15, hi = lane >> 4;
  typedef __attribute__((ext_vector_type(8))) __bf16 v8bf;
  v8bf lo8 = *(const v8bf*)(&ldsB[n][k0 + hi * 8]), hi8 = *(const v8bf*)(&ldsB[n][k0 + 16 + hi * 8]);
  return __builtin_shufflevector(lo8, hi8, 0,1,2,3,4,5,6,7,8,9,10,11,12,13,14,15);
}

#define WMMA_BF16(a, b, c) \
  __builtin_amdgcn_wmma_f32_16x16x32_bf16(false, (a), false, (b), (short)0, (c), false, false)

// ---------------------------------------------------------------------------
// Kernel 0: weight conversion fp32 -> bf16, zero the colsum accumulator.
// ---------------------------------------------------------------------------
__global__ void k_prep(const float* __restrict__ W1, const float* __restrict__ W2,
                       bf16_t* __restrict__ W1bf, bf16_t* __restrict__ W2bf,
                       float* __restrict__ colsum) {
  const int i = blockIdx.x * blockDim.x + threadIdx.x;
  if (i < HID * EMB) W1bf[i] = to_bf16(W1[i]);
  if (i < HID * HID) W2bf[i] = to_bf16(W2[i]);
  if (i < BN)        colsum[i] = 0.0f;
}

// ---------------------------------------------------------------------------
// Kernel 1: embedding gather + MLP encoder (2 WMMA GEMMs) + row norms.
//   hidden[b,n,:]  (f32)   and   norm[b,n,:] = hidden/max(||hidden||,1e-12) (bf16)
// One wave per 16-row tile of the flattened [B*N] rows; 2 waves per block.
// ---------------------------------------------------------------------------
__global__ __launch_bounds__(64) void k_encode(
    const int* __restrict__ tokens, const float* __restrict__ emb_table,
    const float* __restrict__ b1, const float* __restrict__ b2,
    const bf16_t* __restrict__ W1bf, const bf16_t* __restrict__ W2bf,
    float* __restrict__ hidden, bf16_t* __restrict__ normbf) {
  __shared__ __align__(32) bf16_t ldsE[2][16][EMB];
  __shared__ __align__(32) bf16_t ldsH[2][16][HID];
  __shared__ float ldsSS[2][16];

  const int wave = threadIdx.x >> 5;
  const int lane = threadIdx.x & 31;
  const int row0 = (blockIdx.x * 2 + wave) * 16;  // flat row over B*N
  const int col  = lane & 15;
  const int hi   = lane >> 4;

  // Stage 16 embedding rows into LDS as bf16 (gather via tokens).
  for (int idx = lane; idx < 16 * EMB; idx += 32) {
    const int r = idx >> 8;          // EMB == 256
    const int c = idx & (EMB - 1);
    const int tok = tokens[row0 + r];
    ldsE[wave][r][c] = to_bf16(emb_table[(size_t)tok * EMB + c]);
  }
  if (lane < 16) ldsSS[wave][lane] = 0.0f;
  __syncthreads();

  // GEMM1: h = relu(emb @ W1^T + b1) -> LDS (bf16)
  for (int jt = 0; jt < HID / 16; ++jt) {
    const int j = jt * 16 + col;
    const float bj = b1[j];
    // warm the next weight tile while this one computes (global_prefetch_b8)
    __builtin_prefetch(W1bf + (size_t)((jt + 1) * 16 + col) * EMB, 0, 0);
    v8f acc;
#pragma unroll
    for (int i = 0; i < 8; ++i) acc[i] = bj;
#pragma unroll 4
    for (int kt = 0; kt < EMB / 32; ++kt) {
      v16bf a = load_a_lds<EMB>(ldsE[wave], kt * 32, lane);
      v16bf b = load_b_global(W1bf, EMB, jt * 16, kt * 32, lane);
      acc = WMMA_BF16(a, b, acc);
    }
#pragma unroll
    for (int i = 0; i < 8; ++i) {
      const float h = acc[i] > 0.0f ? acc[i] : 0.0f;
      ldsH[wave][i + hi * 8][j] = to_bf16(h);
    }
  }
  __syncthreads();

  // GEMM2: hidden = h @ W2^T + b2 -> global f32, accumulate sum of squares.
  for (int jt = 0; jt < HID / 16; ++jt) {
    const int j = jt * 16 + col;
    const float bj = b2[j];
    __builtin_prefetch(W2bf + (size_t)((jt + 1) * 16 + col) * HID, 0, 0);
    v8f acc;
#pragma unroll
    for (int i = 0; i < 8; ++i) acc[i] = bj;
#pragma unroll 4
    for (int kt = 0; kt < HID / 32; ++kt) {
      v16bf a = load_a_lds<HID>(ldsH[wave], kt * 32, lane);
      v16bf b = load_b_global(W2bf, HID, jt * 16, kt * 32, lane);
      acc = WMMA_BF16(a, b, acc);
    }
#pragma unroll
    for (int i = 0; i < 8; ++i) {
      const int m = i + hi * 8;
      const float v = acc[i];
      hidden[(size_t)(row0 + m) * HID + j] = v;
      atomicAdd(&ldsSS[wave][m], v * v);   // ds_add_f32
    }
  }
  __syncthreads();
  // Ensure our wave's hidden stores are complete before cross-lane re-read.
  asm volatile("s_wait_storecnt 0" ::: "memory");

  // Normalize: norm = hidden / max(||hidden||, 1e-12), stored row-major bf16.
  for (int idx = lane; idx < 16 * HID; idx += 32) {
    const int r = idx >> 9;          // HID == 512
    const int c = idx & (HID - 1);
    const float nrm = sqrtf(ldsSS[wave][r]);
    const float scale = 1.0f / fmaxf(nrm, 1e-12f);
    const float v = hidden[(size_t)(row0 + r) * HID + c];
    normbf[(size_t)(row0 + r) * HID + c] = to_bf16(v * scale);
  }
}

// ---------------------------------------------------------------------------
// Kernel 2: colsum[b,m] = sum_n softplus(norm[b,n,:] . norm[b,m,:])
// Block per (b, mTile of 16 columns). The 16 m-rows of norm are staged in LDS
// (via the Tensor Data Mover when available) and reused by all 8 waves, which
// stripe over the 128 n-tiles. Each wave WMMA-accumulates a 16x16 sim tile
// over K=512, applies softplus in-register, reduces over n and atomically
// adds into colsum.
// ---------------------------------------------------------------------------
__global__ __launch_bounds__(256) void k_sim(const bf16_t* __restrict__ normbf,
                                             float* __restrict__ colsum) {
  __shared__ __align__(32) bf16_t ldsB[16][HID];

  const int b  = blockIdx.x >> 7;    // 128 m-tiles per batch
  const int mt = blockIdx.x & 127;
  const int m0 = mt * 16;
  const bf16_t* nb = normbf + (size_t)b * NSEQ * HID;

#if __has_builtin(__builtin_amdgcn_tensor_load_to_lds)
  // --- TDM staging: one DMA descriptor (D#) moves the 16x1024B tile. -------
  if (threadIdx.x < 32) {            // one wave issues the tensor DMA
    const unsigned long long ga = (unsigned long long)(uintptr_t)(nb + (size_t)m0 * HID);
    const unsigned lds_addr = (unsigned)(uintptr_t)(&ldsB[0][0]);
    // D# group0: count=1, lds_addr, 57-bit global_addr, type=2 ("image")
    v4u g0;
    g0[0] = 1u;
    g0[1] = lds_addr;
    g0[2] = (unsigned)(ga & 0xFFFFFFFFu);
    g0[3] = (unsigned)((ga >> 32) & 0x01FFFFFFu) | 0x80000000u;  // type=2
    // D# group1: data_size=3 (8B elems), tensor_dim0=128, tensor_dim1=NSEQ,
    //            tile_dim0=128, tile_dim1=16, tensor_dim0_stride=128 (=1024B)
    v8i g1;
    g1[0] = (3 << 16);               // workgroup_mask=0 | data_size=8B
    g1[1] = (128 << 16);             // tensor_dim0[15:0] << 16 (abar addr=0)
    g1[2] = (int)((NSEQ & 0xFFFF) << 16);  // tensor_dim0 hi=0 | tensor_dim1 lo
    g1[3] = (128 << 16);             // tensor_dim1 hi=0 | tile_dim0=128
    g1[4] = 16;                      // tile_dim1=16, tile_dim2=0
    g1[5] = 128;                     // tensor_dim0_stride lo (8B units)
    g1[6] = 0;                       // stride hi / tensor_dim1_stride lo
    g1[7] = 0;
    v4i z4 = {};                     // groups 2/3 unused (2D tensor)
#if __clang_major__ >= 23
    v8i z8 = {};
    __builtin_amdgcn_tensor_load_to_lds(g0, g1, z4, z4, z8, 0);
#else
    __builtin_amdgcn_tensor_load_to_lds(g0, g1, z4, z4, 0);
#endif
#if __has_builtin(__builtin_amdgcn_s_wait_tensorcnt)
    __builtin_amdgcn_s_wait_tensorcnt(0);
#else
    asm volatile("s_wait_tensorcnt 0x0" ::: "memory");
#endif
  }
#else
  // --- Fallback staging: cooperative copy of the 16 norm rows into LDS. ----
  for (int idx = threadIdx.x; idx < 16 * (HID / 2); idx += 256) {
    const int r  = idx >> 8;         // HID/2 == 256 u32 per row
    const int c2 = idx & 255;
    ((unsigned int*)&ldsB[r][0])[c2] =
        ((const unsigned int*)(nb + (size_t)(m0 + r) * HID))[c2];
  }
#endif
  __syncthreads();

  const int wave = threadIdx.x >> 5;
  const int lane = threadIdx.x & 31;
  const int colm = lane & 15;

  float local = 0.0f;
  for (int nt = wave; nt < NSEQ / 16; nt += 8) {
    v8f acc = {};
#pragma unroll 4
    for (int kt = 0; kt < HID / 32; ++kt) {
      v16bf a  = load_a_global(nb, HID, nt * 16, kt * 32, lane);
      v16bf bf = load_b_lds(ldsB, kt * 32, lane);
      acc = WMMA_BF16(a, bf, acc);
    }
#pragma unroll
    for (int i = 0; i < 8; ++i) {
      const float x = acc[i];                       // |x| <= 1, no overflow
      local += (x > 20.0f) ? x : log1pf(__expf(x)); // softplus
    }
  }
  atomicAdd(&colsum[(size_t)b * NSEQ + m0 + colm], local);  // global_atomic_add_f32
}

// ---------------------------------------------------------------------------
// Kernel 3: witness[b,:] = (1/N) * colsum[b,:] @ hidden[b,:,:]  (to LDS),
// then out[b,g] = witness[b,:] . Wp[g,:] + bp[g]. One block per batch.
// ---------------------------------------------------------------------------
__global__ __launch_bounds__(512) void k_out(const float* __restrict__ hidden,
                                             const float* __restrict__ colsum,
                                             const float* __restrict__ Wp,
                                             const float* __restrict__ bp,
                                             float* __restrict__ out) {
  __shared__ float wit[HID];
  const int b = blockIdx.x;
  const int h = threadIdx.x;

  const float* hb = hidden + (size_t)b * NSEQ * HID;
  const float* cs = colsum + (size_t)b * NSEQ;
  float acc = 0.0f;
  for (int m = 0; m < NSEQ; ++m) acc += cs[m] * hb[(size_t)m * HID + h];
  wit[h] = acc * (1.0f / (float)NSEQ);
  __syncthreads();

  const int g = threadIdx.x;
  float o = bp[g];
  const float* wrow = Wp + (size_t)g * HID;
#pragma unroll 8
  for (int k = 0; k < HID; ++k) o += wit[k] * wrow[k];
  out[(size_t)b * HID + g] = o;
}

// ---------------------------------------------------------------------------
extern "C" void kernel_launch(void* const* d_in, const int* in_sizes, int n_in,
                              void* d_out, int out_size, void* d_ws, size_t ws_size,
                              hipStream_t stream) {
  (void)in_sizes; (void)n_in; (void)out_size; (void)ws_size;

  const int*   tokens = (const int*)d_in[0];
  const float* emb    = (const float*)d_in[1];
  const float* W1     = (const float*)d_in[2];
  const float* b1     = (const float*)d_in[3];
  const float* W2     = (const float*)d_in[4];
  const float* b2     = (const float*)d_in[5];
  const float* Wp     = (const float*)d_in[6];
  const float* bp     = (const float*)d_in[7];
  float* out = (float*)d_out;

  char* ws = (char*)d_ws;
  size_t off = 0;
  auto alloc = [&](size_t bytes) -> void* {
    void* p = (void*)(ws + off);
    off += (bytes + 255) & ~(size_t)255;
    return p;
  };
  bf16_t* W1bf   = (bf16_t*)alloc((size_t)HID * EMB * sizeof(bf16_t));
  bf16_t* W2bf   = (bf16_t*)alloc((size_t)HID * HID * sizeof(bf16_t));
  float*  hidden = (float*) alloc((size_t)BN * HID * sizeof(float));
  bf16_t* normbf = (bf16_t*)alloc((size_t)BN * HID * sizeof(bf16_t));
  float*  colsum = (float*) alloc((size_t)BN * sizeof(float));

  // 0) weights -> bf16, zero colsum (covers max(HID*HID, BN) elements)
  k_prep<<<(HID * HID + 255) / 256, 256, 0, stream>>>(W1, W2, W1bf, W2bf, colsum);

  // 1) embed + encoder MLP + norms: 512 blocks x 2 waves x 16 rows = 16384 rows
  k_encode<<<BN / 32, 64, 0, stream>>>(tokens, emb, b1, b2, W1bf, W2bf, hidden, normbf);

  // 2) sim + softplus + column sums: block per (b, mTile)
  k_sim<<<BATCH * (NSEQ / 16), 256, 0, stream>>>(normbf, colsum);

  // 3) witness + projection
  k_out<<<BATCH, 512, 0, stream>>>(hidden, colsum, Wp, bp, out);
}
